// SeqFinal_67456756351399
// MI455X (gfx1250) — matrix operand order repaired and checked by reading, not compiled, not run on hardware
//
#include <hip/hip_runtime.h>
#include <hip/hip_bf16.h>
#include <math.h>

// ---------------------------------------------------------------------------
// MI455X (gfx1250) pairwise-MLP, v3.
//
//   a = x @ Wp.T + bp                          [768, 64]
//   comp(i,j) = [a_i, a_j, |a_i-a_j|, a_i*a_j] (256)
//   h   = gelu(comp @ W1.T + b1)               [768,768,512]
//   out = LN(h @ W2.T + b2) * gamma + beta     [768,768,64]
//
// v3 = v2 (all weights LDS-resident in the 320KB WGP LDS; transposed
// formulation: weights = WMMA A operand, comp/h = B operand; lane-half-swap
// shuffle transpose between layers) plus scheduling fixes:
//  * layer-1 K loop software-pipelined: ds_load_b128 for chunk kc+1 issue
//    before the WMMAs of chunk kc (removes per-WMMA s_wait_dscnt stalls)
//  * W2 A-fragments preloaded before the GELU section so their LDS latency
//    hides under the erf VALU chains (VALU co-executes with XDL WMMA)
// ---------------------------------------------------------------------------

typedef __attribute__((ext_vector_type(16))) _Float16 v16h;
typedef __attribute__((ext_vector_type(8)))  float    v8f;

#define IN_DIM  1280
#define NPTS    768
#define PROJ    64
#define COMP_IN 256
#define HID     512
#define OUT_CH  64
#define LN_EPS  1e-5f

#define W2_OFF_H 131072          // halves: W2 fragments start here in LDS
#define SMEM_H   163840          // 327680 bytes = 320 KB
#define FRAG_H   512             // halves per fragment (32 lanes x 16)

__device__ __forceinline__ float gelu_exact(float x) {
    return 0.5f * x * (1.0f + erff(x * 0.70710678118654752f));
}

// ---------------- projection: a = x @ Wp.T + bp  (tiny: 126 MFLOP) ---------
__global__ void proj_kernel(const float* __restrict__ x,
                            const float* __restrict__ Wp,
                            const float* __restrict__ bp,
                            float* __restrict__ a) {
    int gid = blockIdx.x * blockDim.x + threadIdx.x;
    int n = gid >> 6;
    int c = gid & 63;
    if (n >= NPTS) return;
    const float* xr = x + (size_t)n * IN_DIM;
    const float* wr = Wp + (size_t)c * IN_DIM;
    float s = bp[c];
    #pragma unroll 4
    for (int k = 0; k < IN_DIM; k += 4) {
        float4 xv = *(const float4*)(xr + k);
        float4 wv = *(const float4*)(wr + k);
        s += xv.x * wv.x + xv.y * wv.y + xv.z * wv.z + xv.w * wv.w;
    }
    a[n * PROJ + c] = s;
}

// ---------------- weight packing into WMMA *A*-fragment layout -------------
// A-matrix 16x32 f16: lane l -> row M = l%16.
//   e = 0..15 -> k = kc*32 + (l/16)*16 + e
// Flat: wpk[(frag*32 + lane)*16 + e].  W1 frag = n1t*8 + kc (n1t 0..31).
__global__ void pack_w1(const float* __restrict__ W1, _Float16* __restrict__ w1p) {
    int gid = blockIdx.x * blockDim.x + threadIdx.x;      // 131072
    int e    = gid & 15;
    int lane = (gid >> 4) & 31;
    int kc   = (gid >> 9) & 7;
    int n1t  = gid >> 12;
    int n  = n1t * 16 + (lane & 15);
    int kl = (lane >> 4) * 16 + e;
    w1p[gid] = (_Float16)W1[(size_t)n * COMP_IN + kc * 32 + kl];
}

// W2 frag = t2*16 + n1c  (t2 0..3, n1c 0..15); row = channel.
__global__ void pack_w2(const float* __restrict__ W2, _Float16* __restrict__ w2p) {
    int gid = blockIdx.x * blockDim.x + threadIdx.x;      // 32768
    int e    = gid & 15;
    int lane = (gid >> 4) & 31;
    int n1c  = (gid >> 9) & 15;
    int t2   = gid >> 13;
    int n  = t2 * 16 + (lane & 15);
    int kl = (lane >> 4) * 16 + e;
    w2p[gid] = (_Float16)W2[(size_t)n * HID + n1c * 32 + kl];
}

// ---------------- main pairwise MLP kernel ---------------------------------
// Block: 512 threads = 16 waves. Block tile: 16 i-rows x 16 j-cols.
// Wave w: i = blockIdx.y*16 + w; 16 pairs (j0..j0+15) = WMMA N dimension.
__global__ __launch_bounds__(512)
void pair_mlp_kernel(const float* __restrict__ a,
                     const _Float16* __restrict__ wpk,   // packed W1 ++ W2
                     const float* __restrict__ b1,
                     const float* __restrict__ b2,
                     const float* __restrict__ gamma,
                     const float* __restrict__ beta,
                     float* __restrict__ out) {
    __shared__ _Float16 smem[SMEM_H];          // 320 KB: all weights, LDS-resident

    const int tid  = threadIdx.x;
    const int wave = tid >> 5;
    const int lane = tid & 31;
    const int bh   = lane >> 4;     // lane half
    const int nl   = lane & 15;     // pair index within tile (WMMA N)

    // ---- cooperative stage: 320 KB global -> LDS (once per workgroup) -----
    {
        const uint4* src = (const uint4*)wpk;
        uint4*       dst = (uint4*)smem;
        #pragma unroll 8
        for (int t = 0; t < 40; ++t)
            dst[t * 512 + tid] = src[t * 512 + tid];
    }
    __syncthreads();

    const int i  = blockIdx.y * 16 + wave;
    const int j0 = blockIdx.x * 16;
    const float* ai = a + (size_t)i * PROJ;
    const float* aj = a + (size_t)(j0 + nl) * PROJ;   // lane's pair column

    // ---- comp^T as 8 B-fragments (K chunks of 32), built in registers -----
    // B layout: lane l -> col n = l%16 (pair); k = kc*32 + (l/16)*16 + e.
    // Each 32-wide chunk lies inside one feature region (64 | chunk grid).
    v16h B[8];
    #pragma unroll
    for (int kc = 0; kc < 8; ++kc) {
        const int r    = kc >> 1;                      // feature region
        const int base = (kc & 1) * 32 + bh * 16;      // in-region offset
        float4 i0 = *(const float4*)(ai + base);
        float4 i1 = *(const float4*)(ai + base + 4);
        float4 i2 = *(const float4*)(ai + base + 8);
        float4 i3 = *(const float4*)(ai + base + 12);
        float4 j0v = *(const float4*)(aj + base);
        float4 j1v = *(const float4*)(aj + base + 4);
        float4 j2v = *(const float4*)(aj + base + 8);
        float4 j3v = *(const float4*)(aj + base + 12);
        float vi[16] = { i0.x,i0.y,i0.z,i0.w, i1.x,i1.y,i1.z,i1.w,
                         i2.x,i2.y,i2.z,i2.w, i3.x,i3.y,i3.z,i3.w };
        float vj[16] = { j0v.x,j0v.y,j0v.z,j0v.w, j1v.x,j1v.y,j1v.z,j1v.w,
                         j2v.x,j2v.y,j2v.z,j2v.w, j3v.x,j3v.y,j3v.z,j3v.w };
        #pragma unroll
        for (int e = 0; e < 16; ++e) {
            float v = (r == 0) ? vi[e]
                    : (r == 1) ? vj[e]
                    : (r == 2) ? fabsf(vi[e] - vj[e])
                               : vi[e] * vj[e];
            B[kc][e] = (_Float16)v;
        }
    }

    // ---- layer-2 accumulators: out^T tiles (M = channel, N = pair) --------
    // D: VGPR v, lane l -> channel c = t*16 + v + 8*bh, pair = l%16.
    v8f oacc[4];
    #pragma unroll
    for (int t = 0; t < 4; ++t) {
        float4 p0 = *(const float4*)(b2 + t * 16 + 8 * bh);
        float4 p1 = *(const float4*)(b2 + t * 16 + 8 * bh + 4);
        oacc[t][0] = p0.x; oacc[t][1] = p0.y; oacc[t][2] = p0.z; oacc[t][3] = p0.w;
        oacc[t][4] = p1.x; oacc[t][5] = p1.y; oacc[t][6] = p1.z; oacc[t][7] = p1.w;
    }

    // ---- hidden dimension, 32 units per iteration -------------------------
    for (int n1c = 0; n1c < 16; ++n1c) {
        // h^T tile bias: M = n1 = n1c*32 + v + 8*bh (+16 for tile1)
        v8f h0, h1;
        {
            const float* bb = b1 + n1c * 32 + 8 * bh;
            float4 q0 = *(const float4*)(bb);
            float4 q1 = *(const float4*)(bb + 4);
            float4 q2 = *(const float4*)(bb + 16);
            float4 q3 = *(const float4*)(bb + 20);
            h0[0]=q0.x; h0[1]=q0.y; h0[2]=q0.z; h0[3]=q0.w;
            h0[4]=q1.x; h0[5]=q1.y; h0[6]=q1.z; h0[7]=q1.w;
            h1[0]=q2.x; h1[1]=q2.y; h1[2]=q2.z; h1[3]=q2.w;
            h1[4]=q3.x; h1[5]=q3.y; h1[6]=q3.z; h1[7]=q3.w;
        }

        // layer 1: h^T = W1 x comp^T; A-fragments double-buffered from LDS.
        // Fragment f at smem + f*FRAG_H + lane*16; tile0 f = n1c*16 + kc,
        // tile1 f = n1c*16 + 8 + kc.
        const _Float16* wb = smem + (size_t)n1c * (16 * FRAG_H) + lane * 16;
        v16h A0 = *(const v16h*)(wb);
        v16h A1 = *(const v16h*)(wb + 8 * FRAG_H);
        #pragma unroll
        for (int kc = 0; kc < 8; ++kc) {
            const int kn = (kc < 7) ? kc + 1 : 7;          // prefetch index
            v16h N0 = *(const v16h*)(wb + kn * FRAG_H);
            v16h N1 = *(const v16h*)(wb + (8 + kn) * FRAG_H);
            h0 = __builtin_amdgcn_wmma_f32_16x16x32_f16(false, A0, false, B[kc],
                                                        (short)0, h0, false, false);
            h1 = __builtin_amdgcn_wmma_f32_16x16x32_f16(false, A1, false, B[kc],
                                                        (short)0, h1, false, false);
            A0 = N0;
            A1 = N1;
        }

        // preload W2 A-fragments now -- latency hides under the GELU VALU work
        v16h Aw2[4];
        #pragma unroll
        for (int t = 0; t < 4; ++t)
            Aw2[t] = *(const v16h*)(smem + W2_OFF_H
                                    + ((t * 16 + n1c) * 32 + lane) * 16);

        // GELU + lane-half swap: D(M=n1,N=pair) -> B-fragment(lane=pair,k=n1)
        //   lanes<16  need k=0..15  (tile0): own v + xor16 partner
        //   lanes>=16 need k=16..31 (tile1): xor16 partner + own v
        v16h hB;
        #pragma unroll
        for (int v = 0; v < 8; ++v) {
            float g0 = gelu_exact(h0[v]);
            float g1 = gelu_exact(h1[v]);
            float g0x = __shfl_xor(g0, 16, 32);
            float g1x = __shfl_xor(g1, 16, 32);
            float lov = bh ? g1x : g0;     // k_local = v
            float hiv = bh ? g1  : g0x;    // k_local = v + 8
            hB[v]     = (_Float16)lov;
            hB[v + 8] = (_Float16)hiv;
        }

        // layer 2: out^T += W2 x hB, K chunk = n1c
        #pragma unroll
        for (int t = 0; t < 4; ++t)
            oacc[t] = __builtin_amdgcn_wmma_f32_16x16x32_f16(false, Aw2[t], false, hB,
                                                             (short)0, oacc[t], false, false);
    }

    // ---- LayerNorm over 64 channels (32 here + 32 in lane^16) + store -----
    float s = 0.0f, q = 0.0f;
    #pragma unroll
    for (int t = 0; t < 4; ++t)
        #pragma unroll
        for (int v = 0; v < 8; ++v) { s += oacc[t][v]; q += oacc[t][v] * oacc[t][v]; }
    s += __shfl_xor(s, 16, 32);
    q += __shfl_xor(q, 16, 32);
    float mu  = s * (1.0f / 64.0f);
    float var = q * (1.0f / 64.0f) - mu * mu;
    float rs  = rsqrtf(var + LN_EPS);

    float* op = out + (((size_t)i * NPTS) + j0 + nl) * OUT_CH;
    #pragma unroll
    for (int t = 0; t < 4; ++t) {
        float4 g0 = *(const float4*)(gamma + t * 16 + 8 * bh);
        float4 g1 = *(const float4*)(gamma + t * 16 + 8 * bh + 4);
        float4 e0 = *(const float4*)(beta  + t * 16 + 8 * bh);
        float4 e1 = *(const float4*)(beta  + t * 16 + 8 * bh + 4);
        float gm[8] = { g0.x,g0.y,g0.z,g0.w, g1.x,g1.y,g1.z,g1.w };
        float bt[8] = { e0.x,e0.y,e0.z,e0.w, e1.x,e1.y,e1.z,e1.w };
        float val[8];
        #pragma unroll
        for (int v = 0; v < 8; ++v)
            val[v] = (oacc[t][v] - mu) * rs * gm[v] + bt[v];
        float4 o0 = make_float4(val[0], val[1], val[2], val[3]);
        float4 o1 = make_float4(val[4], val[5], val[6], val[7]);
        *(float4*)(op + t * 16 + 8 * bh)     = o0;   // global_store_b128
        *(float4*)(op + t * 16 + 8 * bh + 4) = o1;
    }
}

// ---------------------------------------------------------------------------
extern "C" void kernel_launch(void* const* d_in, const int* in_sizes, int n_in,
                              void* d_out, int out_size, void* d_ws, size_t ws_size,
                              hipStream_t stream) {
    const float* x     = (const float*)d_in[0];
    const float* Wp    = (const float*)d_in[1];
    const float* bp    = (const float*)d_in[2];
    const float* W1    = (const float*)d_in[3];
    const float* b1    = (const float*)d_in[4];
    const float* W2    = (const float*)d_in[5];
    const float* b2    = (const float*)d_in[6];
    const float* gamma = (const float*)d_in[7];
    const float* beta  = (const float*)d_in[8];
    float* out = (float*)d_out;

    char* ws = (char*)d_ws;
    float*    a_proj = (float*)ws;                       // 768*64*4  = 196608 B
    _Float16* w1p    = (_Float16*)(ws + 196608);         // 512*256*2 = 262144 B
    // w2p sits contiguously after w1p: one 320 KB block staged to LDS

    proj_kernel<<<192, 256, 0, stream>>>(x, Wp, bp, a_proj);
    pack_w1<<<512, 256, 0, stream>>>(W1, w1p);
    pack_w2<<<128, 256, 0, stream>>>(W2, w1p + W2_OFF_H);

    dim3 grid(NPTS / 16, NPTS / 16);                     // 48 x 48 workgroups
    pair_mlp_kernel<<<grid, 512, 0, stream>>>(a_proj, w1p,
                                              b1, b2, gamma, beta, out);
}
